// TransformerDecoderLayer_3255585210743
// MI455X (gfx1250) — matrix operand order repaired and checked by reading, not compiled, not run on hardware
//
#include <hip/hip_runtime.h>
#include <cstdint>
#include <cstddef>

// ---------------------------------------------------------------------------
// Types for CDNA5 WMMA: v_wmma_f32_16x16x32_bf16
// ---------------------------------------------------------------------------
typedef __attribute__((ext_vector_type(16))) __bf16 bf16x16;
typedef __attribute__((ext_vector_type(8)))  float  floatx8;
typedef __attribute__((ext_vector_type(4)))  int    v4i;

struct __align__(16) U4 { unsigned int x, y, z, w; };
union Frag { bf16x16 v; U4 u[2]; };

// CDNA5 async global->LDS path (ASYNCcnt), guarded so compile never regresses.
#if defined(__has_builtin)
#if __has_builtin(__builtin_amdgcn_global_load_async_to_lds_b128) && \
    __has_builtin(__builtin_amdgcn_s_wait_asynccnt)
#define USE_ASYNC_LDS 1
#endif
#endif
#ifndef USE_ASYNC_LDS
#define USE_ASYNC_LDS 0
#endif

__device__ __forceinline__ void async_copy16(const unsigned short* g,
                                             unsigned short* l) {
#if USE_ASYNC_LDS
  __builtin_amdgcn_global_load_async_to_lds_b128(
      (__attribute__((address_space(1))) v4i*)(uintptr_t)g,
      (__attribute__((address_space(3))) v4i*)l, 0, 0);
#else
  *(U4*)l = *(const U4*)g;
#endif
}

__device__ __forceinline__ void async_fence() {
#if USE_ASYNC_LDS
  __builtin_amdgcn_s_wait_asynccnt(0);
#endif
}

__device__ __forceinline__ unsigned short f32_to_bf16(float f) {
  unsigned int u = __float_as_uint(f);
  u += 0x7FFFu + ((u >> 16) & 1u);          // round-to-nearest-even
  return (unsigned short)(u >> 16);
}

// ---------------------------------------------------------------------------
// f32 -> bf16 convert
// ---------------------------------------------------------------------------
__global__ void cvt_f32_bf16_kernel(const float* __restrict__ in,
                                    unsigned short* __restrict__ out, int n) {
  int i = blockIdx.x * blockDim.x + threadIdx.x;
  if (i < n) out[i] = f32_to_bf16(in[i]);
}

// ---------------------------------------------------------------------------
// LayerNorm over D=512, one row per block (256 threads, 2 elems/thread).
// ---------------------------------------------------------------------------
__global__ __launch_bounds__(256) void ln_bf16_kernel(
    const float* __restrict__ x, const float* __restrict__ g,
    const float* __restrict__ b, unsigned short* __restrict__ out) {
  constexpr int D = 512;
  __shared__ float red[16];
  int row = blockIdx.x;
  int t = threadIdx.x;
  const float* xr = x + (size_t)row * D;
  float v0 = xr[t], v1 = xr[t + 256];
  float s  = v0 + v1;
  float s2 = v0 * v0 + v1 * v1;
#pragma unroll
  for (int off = 16; off >= 1; off >>= 1) {
    s  += __shfl_xor(s,  off, 32);
    s2 += __shfl_xor(s2, off, 32);
  }
  int wave = t >> 5, lane = t & 31;
  if (lane == 0) { red[wave] = s; red[8 + wave] = s2; }
  __syncthreads();
  float ts = 0.f, ts2 = 0.f;
#pragma unroll
  for (int i = 0; i < 8; ++i) { ts += red[i]; ts2 += red[8 + i]; }
  float mu  = ts * (1.0f / D);
  float var = ts2 * (1.0f / D) - mu * mu;
  float inv = rsqrtf(var + 1e-5f);
  out[(size_t)row * D + t]       = f32_to_bf16((v0 - mu) * inv * g[t]       + b[t]);
  out[(size_t)row * D + t + 256] = f32_to_bf16((v1 - mu) * inv * g[t + 256] + b[t + 256]);
}

// ---------------------------------------------------------------------------
// Tiled bf16 GEMM, f32 accumulate. Block 256 thr = 8 waves (4x2).
// Block tile 128x64, BK=64 (2 WMMA K-steps per barrier pair -> 8 WMMAs/iter).
// A tile staged via CDNA5 async global->LDS; B staged transposed (scatter).
// ---------------------------------------------------------------------------
template <int RELU, int RESID, int OUTF, int OUTB>
__global__ __launch_bounds__(256) void gemm_bf16_kernel(
    const unsigned short* __restrict__ A,   // M x K bf16
    const unsigned short* __restrict__ B,   // K x N bf16
    const float* __restrict__ bias,         // N
    const float* __restrict__ resid,        // M x N f32
    float* __restrict__ outF,               // M x N f32
    unsigned short* __restrict__ outB,      // M x N bf16
    int N, int K) {
  constexpr int BM = 128, BK = 64;
  __shared__ __align__(16) unsigned short As[BM][BK + 8];  // 128 x 72 (row major)
  __shared__ __align__(16) unsigned short Bt[64][BK + 8];  // 64 x 72  (transposed)

  int tid  = threadIdx.x;
  int wave = tid >> 5, lane = tid & 31;
  int wm = wave >> 1, wn = wave & 1;
  int lr = lane & 15, hi = lane >> 4;
  int bm = blockIdx.y * BM, bn = blockIdx.x * 64;

  floatx8 acc[2][2];
#pragma unroll
  for (int i = 0; i < 2; ++i)
#pragma unroll
    for (int j = 0; j < 2; ++j)
      acc[i][j] = (floatx8){0.f, 0.f, 0.f, 0.f, 0.f, 0.f, 0.f, 0.f};

  for (int k0 = 0; k0 < K; k0 += BK) {
    // A tile 128x64: 1024 x 16B, 4 per thread, async direct-to-LDS.
#pragma unroll
    for (int i = 0; i < 4; ++i) {
      int idx = tid + i * 256;
      int r = idx >> 3;
      int c = (idx & 7) * 8;
      async_copy16(A + (size_t)(bm + r) * K + (k0 + c), &As[r][c]);
    }
    // B tile 64x64 transposed into Bt: 2x 16B loads + 16 scatter stores.
#pragma unroll
    for (int i = 0; i < 2; ++i) {
      int idx = tid + i * 256;
      int r = idx >> 3;               // k row
      int c = (idx & 7) * 8;          // n base
      const unsigned short* src = B + (size_t)(k0 + r) * N + (bn + c);
      U4 val = *(const U4*)src;
      if (k0 + BK < K) __builtin_prefetch(src + (size_t)BK * N, 0, 3);
      const unsigned short* vs = (const unsigned short*)&val;
#pragma unroll
      for (int j = 0; j < 8; ++j) Bt[c + j][r] = vs[j];
    }
    async_fence();
    __syncthreads();

#pragma unroll
    for (int s = 0; s < 2; ++s) {     // two K=32 WMMA steps
      Frag a[2], bfr[2];
#pragma unroll
      for (int i = 0; i < 2; ++i) {
        int row = wm * 32 + i * 16 + lr;
        a[i].u[0] = *(const U4*)(&As[row][s * 32 + hi * 8]);
        a[i].u[1] = *(const U4*)(&As[row][s * 32 + 16 + hi * 8]);
      }
#pragma unroll
      for (int j = 0; j < 2; ++j) {
        int col = wn * 32 + j * 16 + lr;
        bfr[j].u[0] = *(const U4*)(&Bt[col][s * 32 + hi * 8]);
        bfr[j].u[1] = *(const U4*)(&Bt[col][s * 32 + 16 + hi * 8]);
      }
#pragma unroll
      for (int i = 0; i < 2; ++i)
#pragma unroll
        for (int j = 0; j < 2; ++j)
          acc[i][j] = __builtin_amdgcn_wmma_f32_16x16x32_bf16(
              false, a[i].v, false, bfr[j].v, (short)0, acc[i][j], false, false);
    }
    __syncthreads();
  }

  // Epilogue: C/D layout lane=N, VGPR r -> M = r + 8*hi.
#pragma unroll
  for (int i = 0; i < 2; ++i)
#pragma unroll
    for (int j = 0; j < 2; ++j) {
      int col = bn + wn * 32 + j * 16 + lr;
      float bv = bias[col];
#pragma unroll
      for (int r = 0; r < 8; ++r) {
        int row = bm + wm * 32 + i * 16 + r + hi * 8;
        float v = acc[i][j][r] + bv;
        if (RELU) v = fmaxf(v, 0.f);
        if (RESID) v += resid[(size_t)row * N + col];
        if (OUTF) outF[(size_t)row * N + col] = v;
        if (OUTB) outB[(size_t)row * N + col] = f32_to_bf16(v);
      }
    }
}

// ---------------------------------------------------------------------------
// Flash attention over 16 chunks of [2048, 64] (reference's flat re-chunking).
// mask indexed [chunk>>3][q_local][k_local]. Block 128 thr = 4 waves, each
// wave owns 16 query rows; 64 keys per iteration (16 WMMAs / iter).
// ---------------------------------------------------------------------------
__global__ __launch_bounds__(128) void attn_kernel(
    const unsigned short* __restrict__ Q,   // [16*2048, 64] bf16
    const unsigned short* __restrict__ Km,
    const unsigned short* __restrict__ Vm,
    const int* __restrict__ mask,           // [2, 2048, 2048]
    unsigned short* __restrict__ O) {       // [16*2048, 64] bf16
  constexpr float SCALE = 0.04419417382415922f;  // 1/sqrt(512)
  __shared__ __align__(16) unsigned short Ks[64][72];      // K tile row-major
  __shared__ __align__(16) unsigned short Vt[64][72];      // V tile transposed
  __shared__ __align__(16) unsigned short Ps[4][16][72];   // per-wave P tile

  int tid  = threadIdx.x;
  int wave = tid >> 5, lane = tid & 31;
  int lr = lane & 15, hi = lane >> 4;
  int chunk = blockIdx.y;
  int batch = chunk >> 3;
  int qbase = blockIdx.x * 64;
  size_t cbase = (size_t)chunk * 2048 * 64;
  const int* mbase = mask + (size_t)batch * 2048 * 2048;

  // Q fragments (A layout), two 32-channel K-chunks, kept in registers.
  Frag qf[2];
  {
    const unsigned short* qp = Q + cbase + (size_t)(qbase + wave * 16 + lr) * 64;
#pragma unroll
    for (int c = 0; c < 2; ++c) {
      qf[c].u[0] = *(const U4*)(qp + c * 32 + hi * 8);
      qf[c].u[1] = *(const U4*)(qp + c * 32 + 16 + hi * 8);
    }
  }

  floatx8 o[4];
#pragma unroll
  for (int ch = 0; ch < 4; ++ch)
    o[ch] = (floatx8){0.f, 0.f, 0.f, 0.f, 0.f, 0.f, 0.f, 0.f};
  float mrow[8], lrow[8];
#pragma unroll
  for (int r = 0; r < 8; ++r) { mrow[r] = -3.0e38f; lrow[r] = 0.f; }

  for (int kb = 0; kb < 2048; kb += 64) {
    // K tile 64x64: async direct-to-LDS (512 x 16B, 4 per thread).
    // V tile 64x64: transposed scatter staging.
#pragma unroll
    for (int i = 0; i < 4; ++i) {
      int idx = tid + i * 128;
      int r = idx >> 3;
      int c = (idx & 7) * 8;
      async_copy16(Km + cbase + (size_t)(kb + r) * 64 + c, &Ks[r][c]);
      U4 vv = *(const U4*)(Vm + cbase + (size_t)(kb + r) * 64 + c);
      const unsigned short* vs = (const unsigned short*)&vv;
#pragma unroll
      for (int j = 0; j < 8; ++j) Vt[c + j][r] = vs[j];
    }
    async_fence();
    __syncthreads();

    // Scores: S = Q[16,64] x K^T[64,64]  (four 16-key tiles, 2 chunks each)
    floatx8 sacc[4];
#pragma unroll
    for (int t = 0; t < 4; ++t) {
      sacc[t] = (floatx8){0.f, 0.f, 0.f, 0.f, 0.f, 0.f, 0.f, 0.f};
#pragma unroll
      for (int c = 0; c < 2; ++c) {
        Frag kf;  // B frag: lane = key (N), elems = channels hi*8+0..7, 16+hi*8+0..7
        int key = t * 16 + lr;
        kf.u[0] = *(const U4*)(&Ks[key][c * 32 + hi * 8]);
        kf.u[1] = *(const U4*)(&Ks[key][c * 32 + 16 + hi * 8]);
        sacc[t] = __builtin_amdgcn_wmma_f32_16x16x32_bf16(
            false, qf[c].v, false, kf.v, (short)0, sacc[t], false, false);
      }
    }

    // Scale + mask, then online softmax (row stats per lane-half).
    float sv[4][8];
#pragma unroll
    for (int t = 0; t < 4; ++t)
#pragma unroll
      for (int r = 0; r < 8; ++r) {
        int ql = qbase + wave * 16 + r + hi * 8;
        int kc = kb + t * 16 + lr;
        int mv = mbase[(size_t)ql * 2048 + kc];
        float s = sacc[t][r] * SCALE;
        sv[t][r] = (mv > 0) ? s : -3.0e38f;
      }
#pragma unroll
    for (int r = 0; r < 8; ++r) {
      float mx = fmaxf(fmaxf(sv[0][r], sv[1][r]), fmaxf(sv[2][r], sv[3][r]));
#pragma unroll
      for (int off = 8; off >= 1; off >>= 1)
        mx = fmaxf(mx, __shfl_xor(mx, off, 32));
      float newm = fmaxf(mrow[r], mx);
      float corr = __expf(mrow[r] - newm);
      float p[4], ps = 0.f;
#pragma unroll
      for (int t = 0; t < 4; ++t) { p[t] = __expf(sv[t][r] - newm); ps += p[t]; }
#pragma unroll
      for (int off = 8; off >= 1; off >>= 1)
        ps += __shfl_xor(ps, off, 32);
      lrow[r] = lrow[r] * corr + ps;
      mrow[r] = newm;
#pragma unroll
      for (int ch = 0; ch < 4; ++ch) o[ch][r] *= corr;
      // C-layout -> LDS (row = r+8*hi, col = t*16+lr)
#pragma unroll
      for (int t = 0; t < 4; ++t)
        Ps[wave][r + hi * 8][t * 16 + lr] = f32_to_bf16(p[t]);
    }
    __syncthreads();

    // O += P[16,64] x V[64,64]   (2 key-chunks x 4 channel tiles)
    Frag pf[2];
#pragma unroll
    for (int c = 0; c < 2; ++c) {
      pf[c].u[0] = *(const U4*)(&Ps[wave][lr][c * 32 + hi * 8]);
      pf[c].u[1] = *(const U4*)(&Ps[wave][lr][c * 32 + 16 + hi * 8]);
    }
#pragma unroll
    for (int ch = 0; ch < 4; ++ch) {
#pragma unroll
      for (int c = 0; c < 2; ++c) {
        Frag vf;  // B frag: lane = channel col, elems = keys c*32+hi*8+0..7, c*32+16+hi*8+0..7
        int col = ch * 16 + lr;
        vf.u[0] = *(const U4*)(&Vt[col][c * 32 + hi * 8]);
        vf.u[1] = *(const U4*)(&Vt[col][c * 32 + 16 + hi * 8]);
        o[ch] = __builtin_amdgcn_wmma_f32_16x16x32_bf16(
            false, pf[c].v, false, vf.v, (short)0, o[ch], false, false);
      }
    }
    __syncthreads();
  }

  // Normalize and write O (bf16, flat layout).
#pragma unroll
  for (int ch = 0; ch < 4; ++ch)
#pragma unroll
    for (int r = 0; r < 8; ++r) {
      int ql = qbase + wave * 16 + r + hi * 8;
      float val = o[ch][r] / lrow[r];
      O[cbase + (size_t)ql * 64 + ch * 16 + lr] = f32_to_bf16(val);
    }
}

// ---------------------------------------------------------------------------
// Orchestration
// ---------------------------------------------------------------------------
extern "C" void kernel_launch(void* const* d_in, const int* in_sizes, int n_in,
                              void* d_out, int out_size, void* d_ws, size_t ws_size,
                              hipStream_t stream) {
  (void)in_sizes; (void)n_in; (void)out_size; (void)ws_size;
  constexpr int Mr = 4096;   // B*S
  constexpr int D  = 512;
  constexpr int H  = 2048;

  const float* x    = (const float*)d_in[0];
  const float* memv = (const float*)d_in[1];
  const int*   mask = (const int*)d_in[2];
  // Weight order: att1 {Wq,Wk,Wv,Wm}, att2 {Wq,Wk,Wv,Wm}, ffn {W1,W2}
  const float* W[10]  = {(const float*)d_in[3],  (const float*)d_in[5],
                         (const float*)d_in[7],  (const float*)d_in[9],
                         (const float*)d_in[11], (const float*)d_in[13],
                         (const float*)d_in[15], (const float*)d_in[17],
                         (const float*)d_in[19], (const float*)d_in[21]};
  const float* Bb[10] = {(const float*)d_in[4],  (const float*)d_in[6],
                         (const float*)d_in[8],  (const float*)d_in[10],
                         (const float*)d_in[12], (const float*)d_in[14],
                         (const float*)d_in[16], (const float*)d_in[18],
                         (const float*)d_in[20], (const float*)d_in[22]};
  const float* lng[3] = {(const float*)d_in[23], (const float*)d_in[25],
                         (const float*)d_in[27]};
  const float* lnb[3] = {(const float*)d_in[24], (const float*)d_in[26],
                         (const float*)d_in[28]};
  const int wsz[10] = {D * D, D * D, D * D, D * D, D * D, D * D, D * D, D * D,
                       D * H, H * D};

  uint8_t* ws = (uint8_t*)d_ws;
  size_t off = 0;
  auto alloc = [&](size_t bytes) -> void* {
    void* p = ws + off;
    off += (bytes + 255) & ~(size_t)255;
    return p;
  };

  unsigned short* wb[10];
  for (int i = 0; i < 10; ++i) wb[i] = (unsigned short*)alloc((size_t)wsz[i] * 2);
  unsigned short* mb   = (unsigned short*)alloc((size_t)Mr * D * 2);
  unsigned short* dbuf = (unsigned short*)alloc((size_t)Mr * D * 2);
  unsigned short* Qb   = (unsigned short*)alloc((size_t)Mr * D * 2);
  unsigned short* Kb   = (unsigned short*)alloc((size_t)Mr * D * 2);
  unsigned short* Vb   = (unsigned short*)alloc((size_t)Mr * D * 2);
  unsigned short* Ob   = (unsigned short*)alloc((size_t)Mr * D * 2);
  float* x1            = (float*)alloc((size_t)Mr * D * 4);
  float* x2            = (float*)alloc((size_t)Mr * D * 4);
  unsigned short* hbuf = (unsigned short*)alloc((size_t)Mr * H * 2);

  auto cvt = [&](const float* src, unsigned short* dst, int n) {
    cvt_f32_bf16_kernel<<<dim3((n + 255) / 256), dim3(256), 0, stream>>>(src, dst, n);
  };
  auto ln = [&](const float* src, int i, unsigned short* dst) {
    ln_bf16_kernel<<<dim3(Mr), dim3(256), 0, stream>>>(src, lng[i], lnb[i], dst);
  };
  // proj: out bf16 only
  auto gemmB = [&](const unsigned short* A, const unsigned short* Bw,
                   const float* bias, unsigned short* outB, int N, int K,
                   int relu) {
    if (relu)
      gemm_bf16_kernel<1, 0, 0, 1><<<dim3(N / 64, Mr / 128), dim3(256), 0, stream>>>(
          A, Bw, bias, nullptr, nullptr, outB, N, K);
    else
      gemm_bf16_kernel<0, 0, 0, 1><<<dim3(N / 64, Mr / 128), dim3(256), 0, stream>>>(
          A, Bw, bias, nullptr, nullptr, outB, N, K);
  };
  // out-proj: residual add, f32 out
  auto gemmR = [&](const unsigned short* A, const unsigned short* Bw,
                   const float* bias, const float* resid, float* outF, int N,
                   int K) {
    gemm_bf16_kernel<0, 1, 1, 0><<<dim3(N / 64, Mr / 128), dim3(256), 0, stream>>>(
        A, Bw, bias, resid, outF, nullptr, N, K);
  };
  auto attn = [&](const unsigned short* Qp, const unsigned short* Kp,
                  const unsigned short* Vp, unsigned short* Op) {
    attn_kernel<<<dim3(32, 16), dim3(128), 0, stream>>>(Qp, Kp, Vp, mask, Op);
  };

  for (int i = 0; i < 10; ++i) cvt(W[i], wb[i], wsz[i]);
  cvt(memv, mb, Mr * D);

  // Block 1: self-attention
  ln(x, 0, dbuf);
  gemmB(dbuf, wb[0], Bb[0], Qb, D, D, 0);
  gemmB(dbuf, wb[1], Bb[1], Kb, D, D, 0);
  gemmB(dbuf, wb[2], Bb[2], Vb, D, D, 0);
  attn(Qb, Kb, Vb, Ob);
  gemmR(Ob, wb[3], Bb[3], x, x1, D, D);

  // Block 2: cross-attention with memory
  ln(x1, 1, dbuf);
  gemmB(dbuf, wb[4], Bb[4], Qb, D, D, 0);
  gemmB(mb,   wb[5], Bb[5], Kb, D, D, 0);
  gemmB(mb,   wb[6], Bb[6], Vb, D, D, 0);
  attn(Qb, Kb, Vb, Ob);
  gemmR(Ob, wb[7], Bb[7], x1, x2, D, D);

  // Block 3: FFN
  ln(x2, 2, dbuf);
  gemmB(dbuf, wb[8], Bb[8], hbuf, H, D, 1);
  gemmR(hbuf, wb[9], Bb[9], x2, (float*)d_out, D, H);
}
